// Attention_36438502539520
// MI455X (gfx1250) — compile-verified, matrix operand order checked by reading, not run on hardware
//
#include <hip/hip_runtime.h>
#include <hip/hip_bf16.h>
#include <math.h>

typedef __attribute__((ext_vector_type(16))) __bf16 v16bf;
typedef __attribute__((ext_vector_type(8)))  float  v8f;

static constexpr int Bb  = 2;
static constexpr int Ss  = 2048;
static constexpr int Dd  = 2048;
static constexpr int Hh  = 16;
static constexpr int DHh = 128;

// ---------- helpers ----------
__device__ inline __bf16 f2bf(float f) {
  union { float f; unsigned u; } a; a.f = f;
  unsigned r = a.u + 0x7FFFu + ((a.u >> 16) & 1u);  // RNE
  unsigned short h = (unsigned short)(r >> 16);
  __bf16 b; __builtin_memcpy(&b, &h, 2);
  return b;
}
__device__ inline float bf2f(__bf16 x) {
  unsigned short h; __builtin_memcpy(&h, &x, 2);
  union { unsigned u; float f; } a; a.u = ((unsigned)h) << 16;
  return a.f;
}

// Load one 16-bit A/B WMMA fragment (16x32 bf16, K contiguous in memory).
// Lane layout (ISA 7.12.2): elements 0..7 = K[8*half .. 8*half+7],
// elements 8..15 = K[16+8*half .. 16+8*half+7]  -> two 16B loads.
__device__ inline v16bf load_frag(const __bf16* row, int half) {
  union { v16bf v; uint4 u[2]; } f;
  const char* p = (const char*)row + half * 16;
  f.u[0] = *(const uint4*)(p);
  f.u[1] = *(const uint4*)(p + 32);
  return f.v;
}

// ---------- fp32 -> bf16 convert (with scale) ----------
__global__ void convert_bf16_kernel(const float* __restrict__ in,
                                    __bf16* __restrict__ out, int n, float scale) {
  int i = blockIdx.x * blockDim.x + threadIdx.x;
  if (i < n) out[i] = f2bf(in[i] * scale);
}

// ---------- bf16 WMMA GEMM: C[M,N] = A[M,K] * B[N,K]^T ----------
// Each wave computes a 32x64 tile (2 A-frags x 4 B-frags -> 8 WMMAs / 12 b128 loads).
// 8 waves arranged 2(M) x 4(N) -> 64x256 block tile.
// mode 0: store bf16 to [b,h,s,dh]   (rows = b*Ss+s, cols = h*DHh+dh)
// mode 1: store bf16 to [b,h,dh,s]   (V transposed for PV fragments)
// mode 2: store fp32 to plain [M,N]
__global__ __launch_bounds__(256) void gemm_bf16_kernel(
    const __bf16* __restrict__ A, const __bf16* __restrict__ Bm,
    __bf16* __restrict__ outb, float* __restrict__ outf,
    int M, int N, int K, int mode, float scale)
{
  const int lane = threadIdx.x & 31;
  const int w    = threadIdx.x >> 5;
  const int m0   = blockIdx.x * 64  + (w & 1) * 32;
  const int n0   = blockIdx.y * 256 + (w >> 1) * 64;
  const int ml   = lane & 15;
  const int half = lane >> 4;

  v8f acc[8];
  v8f zero = {};
  for (int t = 0; t < 8; ++t) acc[t] = zero;

  const __bf16* arow0 = A  + (size_t)(m0 + ml) * K;
  const __bf16* arow1 = arow0 + (size_t)16 * K;
  const __bf16* brow  = Bm + (size_t)(n0 + ml) * K;

  for (int kk = 0; kk < K; kk += 32) {
    v16bf a0 = load_frag(arow0 + kk, half);
    v16bf a1 = load_frag(arow1 + kk, half);
    v16bf b0 = load_frag(brow + kk, half);
    v16bf b1 = load_frag(brow + (size_t)16 * K + kk, half);
    v16bf b2 = load_frag(brow + (size_t)32 * K + kk, half);
    v16bf b3 = load_frag(brow + (size_t)48 * K + kk, half);
    __builtin_prefetch(arow0 + kk + 32, 0, 1);   // global_prefetch_b8 on next A tile
    acc[0] = __builtin_amdgcn_wmma_f32_16x16x32_bf16(false, a0, false, b0, (short)0, acc[0], false, false);
    acc[1] = __builtin_amdgcn_wmma_f32_16x16x32_bf16(false, a0, false, b1, (short)0, acc[1], false, false);
    acc[2] = __builtin_amdgcn_wmma_f32_16x16x32_bf16(false, a0, false, b2, (short)0, acc[2], false, false);
    acc[3] = __builtin_amdgcn_wmma_f32_16x16x32_bf16(false, a0, false, b3, (short)0, acc[3], false, false);
    acc[4] = __builtin_amdgcn_wmma_f32_16x16x32_bf16(false, a1, false, b0, (short)0, acc[4], false, false);
    acc[5] = __builtin_amdgcn_wmma_f32_16x16x32_bf16(false, a1, false, b1, (short)0, acc[5], false, false);
    acc[6] = __builtin_amdgcn_wmma_f32_16x16x32_bf16(false, a1, false, b2, (short)0, acc[6], false, false);
    acc[7] = __builtin_amdgcn_wmma_f32_16x16x32_bf16(false, a1, false, b3, (short)0, acc[7], false, false);
  }

  for (int i = 0; i < 2; ++i) {
    for (int t = 0; t < 4; ++t) {
      const v8f a = acc[i * 4 + t];
      for (int r = 0; r < 8; ++r) {
        const int m = m0 + 16 * i + r + 8 * half;   // C/D row (ISA layout)
        const int n = n0 + 16 * t + ml;             // C/D col
        const float val = a[r] * scale;
        if (mode == 2) {
          outf[(size_t)m * N + n] = val;
        } else {
          const int b  = m / Ss;
          const int s  = m - b * Ss;
          const int hh = n / DHh;
          const int dh = n - hh * DHh;
          size_t idx;
          if (mode == 0) idx = (((size_t)b * Hh + hh) * Ss + s) * DHh + dh;
          else           idx = (((size_t)b * Hh + hh) * DHh + dh) * (size_t)Ss + s;
          outb[idx] = f2bf(val);
        }
      }
    }
  }
}

// ---------- RoPE (in place, on bf16 Q and K in [b,h,s,dh]) ----------
__global__ void rope_kernel(__bf16* __restrict__ Qb, __bf16* __restrict__ Kb, int total) {
  int i = blockIdx.x * blockDim.x + threadIdx.x;
  if (i >= total) return;
  const int j  = i & 63;              // freq index 0..63
  const int s  = (i >> 6) & (Ss - 1); // position
  const int bh = i >> 17;             // b*H + h
  const float inv = __expf(-(float)j * (9.210340371976184f / 64.0f)); // 10000^(-2j/128)
  const float ang = (float)s * inv;
  float si, co;
  __sincosf(ang, &si, &co);
  const size_t base = ((size_t)bh * Ss + s) * DHh;
  float x1 = bf2f(Qb[base + j]), x2 = bf2f(Qb[base + j + 64]);
  Qb[base + j]      = f2bf(x1 * co - x2 * si);
  Qb[base + j + 64] = f2bf(x2 * co + x1 * si);
  x1 = bf2f(Kb[base + j]); x2 = bf2f(Kb[base + j + 64]);
  Kb[base + j]      = f2bf(x1 * co - x2 * si);
  Kb[base + j + 64] = f2bf(x2 * co + x1 * si);
}

// ---------- flash attention: one wave per (b, h, 16-query tile) ----------
__global__ __launch_bounds__(256) void flash_attn_kernel(
    const __bf16* __restrict__ Qb, const __bf16* __restrict__ Kb,
    const __bf16* __restrict__ Vt, __bf16* __restrict__ attn)
{
  __shared__ __bf16 Plds[8][16][40];   // per-wave 16x32 P tile (row stride 40 -> 80B, 16B aligned)

  const int lane = threadIdx.x & 31;
  const int w    = threadIdx.x >> 5;
  const int wid  = blockIdx.x * 8 + w;
  const int NQ   = Ss / 16;
  const int qt   = wid % NQ;
  const int h    = (wid / NQ) % Hh;
  const int b    = wid / (NQ * Hh);
  const int ml   = lane & 15;
  const int half = lane >> 4;

  const __bf16* Qbase = Qb + (((size_t)b * Hh + h) * Ss + (size_t)qt * 16) * DHh;
  const __bf16* Kbase = Kb + (((size_t)b * Hh + h) * Ss) * DHh;
  const __bf16* Vbase = Vt + (((size_t)b * Hh + h) * DHh) * (size_t)Ss;

  // Q fragments for the whole 16x128 tile (4 chunks of K=32)
  v16bf qf[4];
  {
    const __bf16* qrow = Qbase + (size_t)ml * DHh;
    for (int c = 0; c < 4; ++c) qf[c] = load_frag(qrow + 32 * c, half);
  }

  v8f acc[8];
  v8f zero = {};
  for (int c = 0; c < 8; ++c) acc[c] = zero;
  float rmax[8], rsum[8];
  for (int r = 0; r < 8; ++r) { rmax[r] = -3.0e38f; rsum[r] = 0.0f; }

  const float sscale = 0.08838834764831845f;   // 1/sqrt(128)
  const int nkb = (qt * 16 + 16 + 31) >> 5;    // causal: key blocks of 32

  for (int kb = 0; kb < nkb; ++kb) {
    const int k0 = kb * 32;
    // --- scores: Q (16x128) x K^T (128x32) -> two 16x16 tiles ---
    v8f s0 = zero, s1 = zero;
    const __bf16* krow0 = Kbase + (size_t)(k0 + ml) * DHh;
    const __bf16* krow1 = krow0 + (size_t)16 * DHh;
    for (int c = 0; c < 4; ++c) {
      v16bf kf0 = load_frag(krow0 + 32 * c, half);
      v16bf kf1 = load_frag(krow1 + 32 * c, half);
      s0 = __builtin_amdgcn_wmma_f32_16x16x32_bf16(false, qf[c], false, kf0, (short)0, s0, false, false);
      s1 = __builtin_amdgcn_wmma_f32_16x16x32_bf16(false, qf[c], false, kf1, (short)0, s1, false, false);
    }
    // --- online softmax over the 32 new columns ---
    const int qg_base = qt * 16 + 8 * half;
    for (int r = 0; r < 8; ++r) {
      const int qg = qg_base + r;
      float x0 = s0[r] * sscale; if (k0 + ml      > qg) x0 = -3.0e38f;   // causal mask
      float x1 = s1[r] * sscale; if (k0 + 16 + ml > qg) x1 = -3.0e38f;
      float mx = fmaxf(x0, x1);
      mx = fmaxf(mx, __shfl_xor(mx, 1, 32));
      mx = fmaxf(mx, __shfl_xor(mx, 2, 32));
      mx = fmaxf(mx, __shfl_xor(mx, 4, 32));
      mx = fmaxf(mx, __shfl_xor(mx, 8, 32));
      const float mnew  = fmaxf(rmax[r], mx);
      const float alpha = __expf(rmax[r] - mnew);
      const float p0 = __expf(x0 - mnew);
      const float p1 = __expf(x1 - mnew);
      float ps = p0 + p1;
      ps += __shfl_xor(ps, 1, 32);
      ps += __shfl_xor(ps, 2, 32);
      ps += __shfl_xor(ps, 4, 32);
      ps += __shfl_xor(ps, 8, 32);
      rsum[r] = rsum[r] * alpha + ps;
      rmax[r] = mnew;
      for (int c = 0; c < 8; ++c) acc[c][r] *= alpha;
      Plds[w][r + 8 * half][ml]      = f2bf(p0);
      Plds[w][r + 8 * half][16 + ml] = f2bf(p1);
    }
    // wave-private LDS round-trip: reshape P (C-layout) into an A-fragment
    asm volatile("s_wait_dscnt 0" ::: "memory");
    v16bf pf = load_frag(&Plds[w][ml][0], half);
    // --- PV: P (16x32) x V (32x128) accumulated over dh chunks ---
    for (int c = 0; c < 8; ++c) {
      const __bf16* vrow = Vbase + (size_t)(16 * c + ml) * Ss + k0;  // Vt[dh, k]
      v16bf vf = load_frag(vrow, half);
      acc[c] = __builtin_amdgcn_wmma_f32_16x16x32_bf16(false, pf, false, vf, (short)0, acc[c], false, false);
    }
  }

  // --- epilogue: divide by row sums, fold 1/sqrt(H*DH), store bf16 attn [b,s,H*DH] ---
  const float oscale = 0.022097086912079608f;  // 1/sqrt(2048)
  for (int c = 0; c < 8; ++c) {
    for (int r = 0; r < 8; ++r) {
      const int sq = qt * 16 + r + 8 * half;
      const int e  = h * DHh + 16 * c + ml;
      const float val = acc[c][r] / rsum[r] * oscale;
      attn[((size_t)b * Ss + sq) * (size_t)Dd + e] = f2bf(val);
    }
  }
}

// ---------- host orchestration ----------
extern "C" void kernel_launch(void* const* d_in, const int* in_sizes, int n_in,
                              void* d_out, int out_size, void* d_ws, size_t ws_size,
                              hipStream_t stream) {
  const float* hidden = (const float*)d_in[0];
  const float* Wq     = (const float*)d_in[3];
  const float* Wk     = (const float*)d_in[4];
  const float* Wv     = (const float*)d_in[5];
  const float* Wo     = (const float*)d_in[6];

  char* ws = (char*)d_ws;
  size_t off = 0;
  auto carve = [&](size_t bytes) -> void* {
    void* p = ws + off;
    off = (off + bytes + 255) & ~(size_t)255;
    return p;
  };

  const size_t MN  = (size_t)Bb * Ss;          // 4096 rows
  const size_t WE  = (size_t)Dd * Dd;          // 2048*2048 weight elems
  const size_t QKV = (size_t)Bb * Hh * Ss * DHh;

  __bf16* xs    = (__bf16*)carve(MN * Dd * 2);
  __bf16* wqb   = (__bf16*)carve(WE * 2);
  __bf16* wkb   = (__bf16*)carve(WE * 2);
  __bf16* wvb   = (__bf16*)carve(WE * 2);
  __bf16* wob   = (__bf16*)carve(WE * 2);
  __bf16* Qbf   = (__bf16*)carve(QKV * 2);
  __bf16* Kbf   = (__bf16*)carve(QKV * 2);
  __bf16* Vtb   = (__bf16*)carve(QKV * 2);
  __bf16* attnb = (__bf16*)carve(MN * Dd * 2);

  const float inv_sqrt_d = 0.022097086912079608f;  // 1/sqrt(2048)

  // 1) convert inputs/weights to bf16 (hidden pre-scaled by 1/sqrt(D))
  {
    int n = (int)(MN * Dd);
    convert_bf16_kernel<<<(n + 255) / 256, 256, 0, stream>>>(hidden, xs, n, inv_sqrt_d);
    int nw = (int)WE;
    convert_bf16_kernel<<<(nw + 255) / 256, 256, 0, stream>>>(Wq, wqb, nw, 1.0f);
    convert_bf16_kernel<<<(nw + 255) / 256, 256, 0, stream>>>(Wk, wkb, nw, 1.0f);
    convert_bf16_kernel<<<(nw + 255) / 256, 256, 0, stream>>>(Wv, wvb, nw, 1.0f);
    convert_bf16_kernel<<<(nw + 255) / 256, 256, 0, stream>>>(Wo, wob, nw, 1.0f);
  }

  // 2) Q/K/V projections (WMMA bf16); 64x256 block tile, 32x64 per wave
  dim3 gg((unsigned)(MN / 64), (unsigned)(Dd / 256));
  gemm_bf16_kernel<<<gg, 256, 0, stream>>>(xs, wqb, Qbf, nullptr, (int)MN, Dd, Dd, 0, 1.0f);
  gemm_bf16_kernel<<<gg, 256, 0, stream>>>(xs, wkb, Kbf, nullptr, (int)MN, Dd, Dd, 0, 1.0f);
  gemm_bf16_kernel<<<gg, 256, 0, stream>>>(xs, wvb, Vtb, nullptr, (int)MN, Dd, Dd, 1, 1.0f);

  // 3) RoPE on Q and K
  {
    int n = Bb * Hh * Ss * 64;
    rope_kernel<<<(n + 255) / 256, 256, 0, stream>>>(Qbf, Kbf, n);
  }

  // 4) flash attention (one wave per (b,h,qtile); 8 waves per block)
  {
    int nwaves = Bb * Hh * (Ss / 16);
    flash_attn_kernel<<<nwaves / 8, 256, 0, stream>>>(Qbf, Kbf, Vtb, attnb);
  }

  // 5) output projection -> fp32 d_out  (attn already scaled by 1/sqrt(H*DH))
  gemm_bf16_kernel<<<gg, 256, 0, stream>>>(attnb, wob, nullptr, (float*)d_out, (int)MN, Dd, Dd, 2, 1.0f);
}
